// ChristoffelComputer_17480516894899
// MI455X (gfx1250) — matrix operand order repaired and checked by reading, not compiled, not run on hardware
//
#include <hip/hip_runtime.h>
#include <hip/hip_bf16.h>

typedef _Float16 half_t;
typedef __attribute__((ext_vector_type(16))) _Float16 v16h;
typedef __attribute__((ext_vector_type(8)))  _Float16 v8h;
typedef __attribute__((ext_vector_type(8)))  float    v8f;

#define DDIM 16
#define HN   512
#define DD   256
#define PTS  16
#define NTHREADS 512

// ---- CDNA5 WMMA: D = A(16x32 f16) x B(32x16 f16) + C(16x16 f32) ----
__device__ __forceinline__ v8f wmma16(v16h a, v16h b, v8f c) {
  return __builtin_amdgcn_wmma_f32_16x16x32_f16(
      /*neg_a=*/false, a, /*neg_b=*/false, b,
      /*c_mod=*/(short)0, c, /*reuse_a=*/false, /*reuse_b=*/false);
}

__device__ __forceinline__ v16h cat8(v8h lo, v8h hi) {
  return __builtin_shufflevector(lo, hi, 0,1,2,3,4,5,6,7,8,9,10,11,12,13,14,15);
}

__device__ __forceinline__ float readlane_f(float v, int l) {
  return __int_as_float(__builtin_amdgcn_readlane(__float_as_int(v), l));
}

__global__ void convert_f32_to_f16(const float* __restrict__ src,
                                   half_t* __restrict__ dst, int n) {
  int i = blockIdx.x * 256 + threadIdx.x;
  if (i < n) dst[i] = (half_t)src[i];
}

// One workgroup = 16 points (16 waves / 512 threads).
__global__ void __launch_bounds__(NTHREADS)
christoffel_kernel(const float* __restrict__ z,
                   const float* __restrict__ W1,
                   const float* __restrict__ b1,
                   const half_t* __restrict__ W2h,
                   const float* __restrict__ b2,
                   float* __restrict__ out)
{
  extern __shared__ char smem[];
  // ---- region A: live during P0/P1 (GEMM phase) ----
  half_t* sW1h  = (half_t*)(smem);             // [16][512] f16   16KB
  half_t* sHch  = (half_t*)(smem + 16384);     // [16][512] f16   16KB
  half_t* sSh   = (half_t*)(smem + 32768);     // [16][512] f16   16KB
  half_t* sUdhA = (half_t*)(smem + 49152);     // [16pt][16p][32] 16KB
  half_t* sUdhB = (half_t*)(smem + 65536);     // double buffer   16KB
  // ---- region A aliases: live during P3 (epilogue) ----
  float*  sGtmp = (float*)(smem);              // [16pt][16][16]  16KB (g spill)
  float*  sGi   = (float*)(smem + 16384);      // [16pt][16][16]  16KB (0.5*g^-1)
  // ---- persistent ----
  float*  sAc   = (float*)(smem + 81920);      // [16pt][16][16]  16KB
  float*  sUd   = (float*)(smem + 98304);      // [16p][16][16]   16KB
  float*  sM    = (float*)(smem + 114688);     // [16p][16][16]   16KB
  float*  sZ    = (float*)(smem + 131072);     // [16][16]        1KB

  const int tid  = threadIdx.x;
  const int lane = tid & 31;
  const int wave = tid >> 5;     // 0..15
  const int hsel = lane >> 4;    // 0/1 : K-half select (wave32 split)
  const int m16  = lane & 15;

  // ---------------- P0: encoder front-end (vectorized VALU) ----------------
  #pragma unroll
  for (int r = 0; r < 16; ++r) {
    int idx = tid + NTHREADS * r;              // 0..8191
    sW1h[idx] = (half_t)W1[idx];
  }
  if (tid < 256) sZ[tid] = z[(size_t)blockIdx.x * 256 + tid];
  __syncthreads();

  {
    const int pt = wave;
    const int j0 = lane * 16;                  // 16 contiguous columns per lane
    float zr[16];
    #pragma unroll
    for (int d = 0; d < DDIM; ++d) zr[d] = sZ[pt * 16 + d];
    float c[16];
    #pragma unroll
    for (int e = 0; e < 16; ++e) c[e] = b1[j0 + e];
    #pragma unroll
    for (int d = 0; d < DDIM; ++d) {
      v8h w0 = *(const v8h*)&sW1h[d * HN + j0];
      v8h w1 = *(const v8h*)&sW1h[d * HN + j0 + 8];
      #pragma unroll
      for (int e = 0; e < 8; ++e) {
        c[e]     += zr[d] * (float)w0[e];
        c[e + 8] += zr[d] * (float)w1[e];
      }
    }
    v8h h0, h1, s0, s1;
    #pragma unroll
    for (int e = 0; e < 8; ++e) {
      float ta = tanhf(c[e]);
      float tb = tanhf(c[e + 8]);
      h0[e] = (half_t)ta;  s0[e] = (half_t)(1.0f - ta * ta);
      h1[e] = (half_t)tb;  s1[e] = (half_t)(1.0f - tb * tb);
    }
    *(v8h*)&sHch[pt * HN + j0]     = h0;
    *(v8h*)&sHch[pt * HN + j0 + 8] = h1;
    *(v8h*)&sSh[pt * HN + j0]      = s0;
    *(v8h*)&sSh[pt * HN + j0 + 8]  = s1;
  }

  // Udh slice builder: thread -> (pt, p, half16); pure v8h packed math.
  const int bpt = tid >> 5;
  const int bp  = (tid >> 1) & 15;
  const int bhh = tid & 1;
  #define BUILD_UDH(buf, kkv)                                                  \
    {                                                                          \
      int jb = (kkv) * 32 + bhh * 16;                                          \
      v8h w0 = *(const v8h*)&sW1h[bp * HN + jb];                               \
      v8h w1 = *(const v8h*)&sW1h[bp * HN + jb + 8];                           \
      v8h q0 = *(const v8h*)&sSh[bpt * HN + jb];                               \
      v8h q1 = *(const v8h*)&sSh[bpt * HN + jb + 8];                           \
      *(v8h*)&(buf)[bpt * HN + bp * 32 + bhh * 16]     = w0 * q0;              \
      *(v8h*)&(buf)[bpt * HN + bp * 32 + bhh * 16 + 8] = w1 * q1;              \
    }

  __syncthreads();
  BUILD_UDH(sUdhA, 0)
  __syncthreads();

  // ---------------- P1: big WMMA GEMMs over K=512 ----------------
  v8f accC; v8f accU[16];
  #pragma unroll
  for (int e = 0; e < 8; ++e) accC[e] = 0.0f;
  #pragma unroll
  for (int p = 0; p < 16; ++p) {
    #pragma unroll
    for (int e = 0; e < 8; ++e) accU[p][e] = 0.0f;
  }

  for (int kk = 0; kk < 16; ++kk) {
    const half_t* cur = (kk & 1) ? sUdhB : sUdhA;
    half_t*       nxt = (kk & 1) ? sUdhA : sUdhB;

    if (kk < 15)  // global_prefetch_b8 of next W2 K-slice
      __builtin_prefetch(W2h + ((size_t)(kk + 1) * 32 + lane) * DD + wave * 16, 0, 1);

    // B fragment: lane%16 = N (16-wide tile), elements = K in table order
    v16h bfr;
    #pragma unroll
    for (int s = 0; s < 16; ++s)
      bfr[s] = W2h[((size_t)kk * 32 + (s < 8 ? 8 * hsel + s : 16 + 8 * hsel + (s - 8))) * DD + wave * 16 + m16];

    { // center: Ac += Hc * W2
      v8h lo = *(const v8h*)&sHch[m16 * HN + kk * 32 + 8 * hsel];
      v8h hi = *(const v8h*)&sHch[m16 * HN + kk * 32 + 16 + 8 * hsel];
      accC = wmma16(cat8(lo, hi), bfr, accC);
    }
    #pragma unroll
    for (int pt = 0; pt < 16; ++pt) {        // 16 points reuse the same B frag
      v8h lo = *(const v8h*)&cur[pt * HN + m16 * 32 + 8 * hsel];
      v8h hi = *(const v8h*)&cur[pt * HN + m16 * 32 + 16 + 8 * hsel];
      accU[pt] = wmma16(cat8(lo, hi), bfr, accU[pt]);
    }
    if (kk < 15) BUILD_UDH(nxt, kk + 1)      // overlap build with consume
    __syncthreads();                          // one barrier per K-step
  }

  // ---------------- P2: spill Ac (+b2) to LDS ----------------
  #pragma unroll
  for (int r = 0; r < 8; ++r) {
    int pt_ = r + 8 * hsel;
    sAc[pt_ * DD + wave * 16 + m16] = accC[r] + b2[wave * 16 + m16];
  }
  __syncthreads();

  // ---------------- P3-pre: wave w inverts g of point w ----------------
  {
    const int w = wave;
    // g = A A^T via WMMA (one extra wmma per point)
    v16h a, b;
    #pragma unroll
    for (int e = 0; e < 16; ++e) { a[e] = (half_t)0.f; b[e] = (half_t)0.f; }
    #pragma unroll
    for (int e = 0; e < 16; ++e)
      a[e] = hsel ? (half_t)0.f : (half_t)sAc[w * DD + m16 * 16 + e];
    if (lane < 16) {
      #pragma unroll
      for (int e = 0; e < 16; ++e)
        b[e] = (half_t)sAc[w * DD + e * 16 + lane];
    }
    v8f gac;
    #pragma unroll
    for (int e = 0; e < 8; ++e) gac[e] = 0.0f;
    gac = wmma16(a, b, gac);
    #pragma unroll
    for (int r = 0; r < 8; ++r)
      sGtmp[w * DD + (r + 8 * hsel) * 16 + m16] = gac[r];
    __syncthreads();

    // Register-resident Gauss-Jordan on [g+I | I]: lane = augmented column,
    // pivot column broadcast via v_readlane; no LDS, no barriers.
    float col[16];
    if (lane < 16) {
      #pragma unroll
      for (int r = 0; r < 16; ++r)
        col[r] = sGtmp[w * DD + r * 16 + lane] + ((r == lane) ? 1.0f : 0.0f);
    } else {
      #pragma unroll
      for (int r = 0; r < 16; ++r)
        col[r] = (r == lane - 16) ? 1.0f : 0.0f;
    }
    #pragma unroll
    for (int k = 0; k < 16; ++k) {
      float rp = 1.0f / readlane_f(col[k], k);       // 1/G[k,k]
      #pragma unroll
      for (int r = 0; r < 16; ++r) {
        if (r == k) continue;
        float fac = readlane_f(col[r], k) * rp;      // G[r,k]/G[k,k]
        col[r] -= fac * col[k];
      }
    }
    #pragma unroll
    for (int r = 0; r < 16; ++r) {                   // fold the 1/2 of Gamma
      float rdr = 0.5f / readlane_f(col[r], r);
      if (lane >= 16) sGi[w * DD + r * 16 + (lane - 16)] = col[r] * rdr;
    }
  }
  __syncthreads();

  // ---------------- P3: per-point epilogue ----------------
  #pragma unroll
  for (int pt = 0; pt < 16; ++pt) {
    // Ud (this point) to LDS: sUd[p][m][n]
    #pragma unroll
    for (int r = 0; r < 8; ++r) {
      int p_ = r + 8 * hsel;
      sUd[p_ * DD + wave * 16 + m16] = accU[pt][r];
    }
    __syncthreads();

    // Step 6: wave p computes M_p = A * Ud_p^T   (16x16x16, K padded to 32)
    {
      const int p = wave;
      v16h a, b;
      #pragma unroll
      for (int e = 0; e < 16; ++e) { a[e] = (half_t)0.f; b[e] = (half_t)0.f; }
      #pragma unroll
      for (int e = 0; e < 16; ++e)
        a[e] = hsel ? (half_t)0.f : (half_t)sAc[pt * DD + m16 * 16 + e];
      if (lane < 16) {
        #pragma unroll
        for (int e = 0; e < 16; ++e)
          b[e] = (half_t)sUd[p * DD + e * 16 + lane];
      }
      v8f Macc;
      #pragma unroll
      for (int e = 0; e < 8; ++e) Macc[e] = 0.0f;
      Macc = wmma16(a, b, Macc);
      #pragma unroll
      for (int r = 0; r < 8; ++r)
        sM[p * DD + (r + 8 * hsel) * 16 + m16] = Macc[r];
    }
    __syncthreads();

    // Step 7: Gamma[k, i=wave, j] = (1/2 g^-1) @ S; dg(a,b,p)=M_p[a,b]+M_p[b,a]
    {
      const int i_ = wave;
      v16h a, b;
      #pragma unroll
      for (int e = 0; e < 16; ++e) { a[e] = (half_t)0.f; b[e] = (half_t)0.f; }
      #pragma unroll
      for (int e = 0; e < 16; ++e)
        a[e] = hsel ? (half_t)0.f : (half_t)sGi[pt * DD + m16 * 16 + e];
      if (lane < 16) {
        const int mm = lane;
        #pragma unroll
        for (int j_ = 0; j_ < 16; ++j_) {
          float s = sM[i_ * DD + j_ * 16 + mm] + sM[i_ * DD + mm * 16 + j_]
                  + sM[j_ * DD + i_ * 16 + mm] + sM[j_ * DD + mm * 16 + i_]
                  - sM[mm * DD + i_ * 16 + j_] - sM[mm * DD + j_ * 16 + i_];
          b[j_] = (half_t)s;
        }
      }
      v8f Gacc;
      #pragma unroll
      for (int e = 0; e < 8; ++e) Gacc[e] = 0.0f;
      Gacc = wmma16(a, b, Gacc);

      size_t ptg = (size_t)blockIdx.x * 16 + pt;
      float* o = out + ptg * 4096 + (size_t)i_ * 16;   // [pt][k][i][j]
      #pragma unroll
      for (int r = 0; r < 8; ++r)
        o[(size_t)(r + 8 * hsel) * DD + m16] = Gacc[r];
    }
    __syncthreads();
  }
}

extern "C" void kernel_launch(void* const* d_in, const int* in_sizes, int n_in,
                              void* d_out, int out_size, void* d_ws, size_t ws_size,
                              hipStream_t stream) {
  const float* z  = (const float*)d_in[0];
  const float* W1 = (const float*)d_in[1];
  const float* b1 = (const float*)d_in[2];
  const float* W2 = (const float*)d_in[3];
  const float* b2 = (const float*)d_in[4];
  float* out = (float*)d_out;
  half_t* W2h = (half_t*)d_ws;              // 512*256 f16 = 256KB scratch

  convert_f32_to_f16<<<(HN * DD + 255) / 256, 256, 0, stream>>>(W2, W2h, HN * DD);

  const int nBlocks = (8 * 1024) / PTS;     // 512 workgroups
  const size_t smemBytes = 132096;          // ~129KB of the 320KB/WGP LDS
  christoffel_kernel<<<nBlocks, NTHREADS, smemBytes, stream>>>(z, W1, b1, W2h, b2, out);

  (void)in_sizes; (void)n_in; (void)out_size; (void)ws_size;
}
